// SoftmaxTaskEncoder_20727512171104
// MI455X (gfx1250) — compile-verified, matrix operand order checked by reading, not run on hardware
//
#include <hip/hip_runtime.h>
#include <hip/hip_bf16.h>
#include <math.h>

typedef __bf16 bf16_t;
typedef __attribute__((ext_vector_type(16))) __bf16 v16bf;
typedef __attribute__((ext_vector_type(8)))  float  v8f;

#define TASK_EMB 1024
#define HID      256
#define N_EMBS   4096
#define CODE_DIM 64
#define BATCH    16384
#define LN_EPS   1e-5f

// padded LDS row strides (elements); 32B-aligned for v16bf loads
#define W2S 272   // 256 + 16
#define W1S 80    // 64 + 16
#define EMS 80    // 64 + 16
#define PSS 80    // 64 + 16

__device__ __forceinline__ float red_max16(float v) {
  v = fmaxf(v, __shfl_xor(v, 1));
  v = fmaxf(v, __shfl_xor(v, 2));
  v = fmaxf(v, __shfl_xor(v, 4));
  v = fmaxf(v, __shfl_xor(v, 8));
  return v;
}
__device__ __forceinline__ float red_sum16(float v) {
  v += __shfl_xor(v, 1);
  v += __shfl_xor(v, 2);
  v += __shfl_xor(v, 4);
  v += __shfl_xor(v, 8);
  return v;
}

__device__ __forceinline__ v8f wmma_bf16(v16bf a, v16bf b, v8f c) {
  return __builtin_amdgcn_wmma_f32_16x16x32_bf16(false, a, false, b, (short)0, c,
                                                 false, false);
}

// CDNA5 async copy: 16B per lane, global -> LDS, tracked by ASYNCcnt (ISA §10/§8)
__device__ __forceinline__ void async_cp16(unsigned lds_byte_addr, const void* g) {
  asm volatile("global_load_async_to_lds_b128 %0, %1, off"
               :: "v"(lds_byte_addr), "v"(g) : "memory");
}
// Wait until at most N async loads remain in flight (in-order completion).
template <int N>
__device__ __forceinline__ void wait_async_le() {
  asm volatile("s_wait_asynccnt %0" :: "i"(N) : "memory");
}
__device__ __forceinline__ unsigned lds_addr(const void* p) {
  return (unsigned)(uintptr_t)p;   // LDS aperture: low 32 bits = LDS byte offset
}

// ---------------------------------------------------------------------------
// Convert f32 (rows x cols, row-major) -> bf16 transposed (cols x rows)
// ---------------------------------------------------------------------------
__global__ void cvt_transpose_bf16(const float* __restrict__ in,
                                   bf16_t* __restrict__ out, int rows, int cols) {
  int i = blockIdx.x * blockDim.x + threadIdx.x;
  int n = rows * cols;
  if (i >= n) return;
  int r = i / cols, c = i % cols;
  out[(long)c * rows + r] = (bf16_t)in[i];
}

// ---------------------------------------------------------------------------
// h = silu(x @ W1 + b1). One wave computes a full 16x256 row-tile (x read
// exactly once). W1^T K-slices double-buffered in LDS via async copies.
// ---------------------------------------------------------------------------
__device__ __forceinline__ void stage_w1(int tid, unsigned dst,
                                         const bf16_t* w1t, int k0) {
#pragma unroll
  for (int i = 0; i < 8; ++i) {           // 256 rows x 128B, 8 segs/thread
    int s = tid + i * 256;
    int row = s >> 3, seg = s & 7;
    async_cp16(dst + row * (W1S * 2) + seg * 16,
               (const char*)w1t + ((long)row * TASK_EMB + k0) * 2 + seg * 16);
  }
}

__global__ __launch_bounds__(256) void gemm1_silu(
    const float* __restrict__ x, const bf16_t* __restrict__ w1t,
    const float* __restrict__ b1, bf16_t* __restrict__ h) {
  __shared__ __align__(32) bf16_t sW1[2][256 * W1S];

  const int tid  = threadIdx.x;
  const int lane = tid & 31;
  const int wslot = tid >> 5;
  const int m0 = (blockIdx.x * 8 + wslot) * 16;
  const int lr = lane & 15;
  const int hi = lane >> 4;
  const int kh = hi * 16;

  v8f acc[16];
#pragma unroll
  for (int nt = 0; nt < 16; ++nt)
    acc[nt] = (v8f){0.f,0.f,0.f,0.f,0.f,0.f,0.f,0.f};

  const float* ap = x + (long)(m0 + lr) * TASK_EMB + kh;

  stage_w1(tid, lds_addr(sW1[0]), w1t, 0);            // prologue: chunk 0
  for (int i = 0; i < 16; ++i) {
    const int k0 = i * 64;
    const bf16_t* sw = sW1[i & 1];
    if (i < 15) {                                     // prefetch next chunk
      stage_w1(tid, lds_addr(sW1[(i + 1) & 1]), w1t, k0 + 64);
      wait_async_le<8>();                             // chunk i landed
    } else {
      wait_async_le<0>();
    }
    __syncthreads();                                  // visible to all waves

#pragma unroll
    for (int ks = 0; ks < 2; ++ks) {
      const float4* af = (const float4*)(ap + k0 + ks * 32);
      float4 f0 = af[0], f1 = af[1], f2 = af[2], f3 = af[3];
      v16bf a;
      a[0]=(bf16_t)f0.x; a[1]=(bf16_t)f0.y; a[2]=(bf16_t)f0.z; a[3]=(bf16_t)f0.w;
      a[4]=(bf16_t)f1.x; a[5]=(bf16_t)f1.y; a[6]=(bf16_t)f1.z; a[7]=(bf16_t)f1.w;
      a[8]=(bf16_t)f2.x; a[9]=(bf16_t)f2.y; a[10]=(bf16_t)f2.z; a[11]=(bf16_t)f2.w;
      a[12]=(bf16_t)f3.x; a[13]=(bf16_t)f3.y; a[14]=(bf16_t)f3.z; a[15]=(bf16_t)f3.w;
#pragma unroll
      for (int nt = 0; nt < 16; ++nt) {
        v16bf b = *(const v16bf*)(sw + (nt * 16 + lr) * W1S + ks * 32 + kh);
        acc[nt] = wmma_bf16(a, b, acc[nt]);
      }
    }
    __syncthreads();                 // chunk i consumed; buffer reusable
  }

#pragma unroll
  for (int nt = 0; nt < 16; ++nt) {
    float bias = b1[nt * 16 + lr];
#pragma unroll
    for (int r = 0; r < 8; ++r) {
      float v = acc[nt][r] + bias;
      float s = v * __builtin_amdgcn_rcpf(1.0f + __expf(-v));   // fast SiLU
      h[(long)(m0 + r + 8 * hi) * HID + nt * 16 + lr] = (bf16_t)s;
    }
  }
}

// ---------------------------------------------------------------------------
// Fused: logits = h @ W2 + b2 ; online softmax ; emb = probs @ E ; LayerNorm.
// W2^T / E^T chunks double-buffered in LDS via async copies (10/thread/chunk).
// ---------------------------------------------------------------------------
__device__ __forceinline__ void stage_chunk(int tid, unsigned dW2, unsigned dEmb,
                                            const bf16_t* w2t,
                                            const bf16_t* embT, int c0) {
#pragma unroll
  for (int i = 0; i < 8; ++i) {           // W2^T: 64 rows x 512B, 8 segs/thread
    int s = tid + i * 256;
    int row = s >> 5, seg = s & 31;
    async_cp16(dW2 + row * (W2S * 2) + seg * 16,
               (const char*)w2t + ((long)(c0 + row) * HID) * 2 + seg * 16);
  }
#pragma unroll
  for (int i = 0; i < 2; ++i) {           // E^T: 64 rows x 128B, 2 segs/thread
    int s = tid + i * 256;
    int row = s >> 3, seg = s & 7;
    async_cp16(dEmb + row * (EMS * 2) + seg * 16,
               (const char*)embT + ((long)row * N_EMBS + c0) * 2 + seg * 16);
  }
}

__global__ __launch_bounds__(256) void fused_softmax_code(
    const bf16_t* __restrict__ h, const bf16_t* __restrict__ w2t,
    const float* __restrict__ b2, const bf16_t* __restrict__ embT,
    const float* __restrict__ gamma, const float* __restrict__ beta,
    float* __restrict__ out) {
  __shared__ __align__(32) bf16_t sW2[2][64 * W2S];
  __shared__ __align__(32) bf16_t sEmb[2][64 * EMS];
  __shared__ __align__(32) bf16_t psm[8][16 * PSS];

  const int tid   = threadIdx.x;
  const int lane  = tid & 31;
  const int wslot = tid >> 5;
  const int m0 = (blockIdx.x * 8 + wslot) * 16;
  const int lr = lane & 15;
  const int hi = lane >> 4;
  const int kh = hi * 16;

  // Preload h A-fragments for K = 0..255 (8 frags, kept in registers)
  v16bf ha[8];
  {
    const bf16_t* hp = h + (long)(m0 + lr) * HID + kh;
#pragma unroll
    for (int ks = 0; ks < 8; ++ks) ha[ks] = *(const v16bf*)(hp + ks * 32);
  }

  v8f acc[4];
#pragma unroll
  for (int nt = 0; nt < 4; ++nt) acc[nt] = (v8f){0.f,0.f,0.f,0.f,0.f,0.f,0.f,0.f};
  float m_[8], l_[8];
#pragma unroll
  for (int r = 0; r < 8; ++r) { m_[r] = -3.0e38f; l_[r] = 0.f; }

  bf16_t* sp = psm[wslot];

  stage_chunk(tid, lds_addr(sW2[0]), lds_addr(sEmb[0]), w2t, embT, 0);
  for (int i = 0; i < 64; ++i) {
    const int c0 = i * 64;
    const bf16_t* sw = sW2[i & 1];
    const bf16_t* se = sEmb[i & 1];
    if (i < 63) {                                     // prefetch next chunk
      stage_chunk(tid, lds_addr(sW2[(i + 1) & 1]), lds_addr(sEmb[(i + 1) & 1]),
                  w2t, embT, c0 + 64);
      wait_async_le<10>();                            // chunk i landed
    } else {
      wait_async_le<0>();
    }
    __syncthreads();

    // ---- logits chunk: 16x64 = 4 n-tiles, K=256 in 8 WMMA steps each ----
    v8f lg[4];
#pragma unroll
    for (int nt = 0; nt < 4; ++nt) {
      v8f t = {0.f,0.f,0.f,0.f,0.f,0.f,0.f,0.f};
      const bf16_t* wp = sw + (nt * 16 + lr) * W2S + kh;
#pragma unroll
      for (int ks = 0; ks < 8; ++ks) {
        v16bf b = *(const v16bf*)(wp + ks * 32);
        t = wmma_bf16(ha[ks], b, t);
      }
      const float bias = b2[c0 + nt * 16 + lr];
#pragma unroll
      for (int r = 0; r < 8; ++r) t[r] += bias;
      lg[nt] = t;
    }

    // ---- online softmax (per-row over the 16-lane halves) ----
#pragma unroll
    for (int r = 0; r < 8; ++r) {
      float cm = fmaxf(fmaxf(lg[0][r], lg[1][r]), fmaxf(lg[2][r], lg[3][r]));
      cm = red_max16(cm);
      float mn = fmaxf(m_[r], cm);
      float sc = __expf(m_[r] - mn);
      m_[r] = mn;
      l_[r] *= sc;
      float ps = 0.f;
#pragma unroll
      for (int nt = 0; nt < 4; ++nt) {
        acc[nt][r] *= sc;
        float p = __expf(lg[nt][r] - mn);
        lg[nt][r] = p;
        ps += p;
      }
      l_[r] += red_sum16(ps);
    }

    // ---- stage probs via LDS: D-layout -> A-layout (per-wave region) ----
#pragma unroll
    for (int nt = 0; nt < 4; ++nt)
#pragma unroll
      for (int r = 0; r < 8; ++r)
        sp[(r + 8 * hi) * PSS + nt * 16 + lr] = (bf16_t)lg[nt][r];
    __builtin_amdgcn_wave_barrier();       // LDS same-wave ops are in-order

    v16bf pa0 = *(const v16bf*)(sp + lr * PSS + kh);
    v16bf pa1 = *(const v16bf*)(sp + lr * PSS + 32 + kh);

    // ---- acc += probs_chunk @ E[c0:c0+64, :] ----
#pragma unroll
    for (int nt = 0; nt < 4; ++nt) {
      const bf16_t* ep = se + (nt * 16 + lr) * EMS + kh;
      v16bf e0 = *(const v16bf*)(ep);
      v16bf e1 = *(const v16bf*)(ep + 32);
      acc[nt] = wmma_bf16(pa0, e0, acc[nt]);
      acc[nt] = wmma_bf16(pa1, e1, acc[nt]);
    }
    __syncthreads();                 // chunk i consumed; buffer reusable
  }

  // ---- normalize by l, LayerNorm over CODE_DIM=64, store f32 ----
  float g_[4], b_[4];
#pragma unroll
  for (int nt = 0; nt < 4; ++nt) {
    g_[nt] = gamma[nt * 16 + lr];
    b_[nt] = beta[nt * 16 + lr];
  }
#pragma unroll
  for (int r = 0; r < 8; ++r) {
    float inv = __builtin_amdgcn_rcpf(l_[r]);
    float e[4], s = 0.f, sq = 0.f;
#pragma unroll
    for (int nt = 0; nt < 4; ++nt) {
      e[nt] = acc[nt][r] * inv;
      s += e[nt];
      sq += e[nt] * e[nt];
    }
    s  = red_sum16(s);
    sq = red_sum16(sq);
    float mean = s * (1.0f / 64.0f);
    float var  = sq * (1.0f / 64.0f) - mean * mean;
    float rstd = __builtin_amdgcn_rsqf(var + LN_EPS);
    float* op = out + (long)(m0 + r + 8 * hi) * CODE_DIM;
#pragma unroll
    for (int nt = 0; nt < 4; ++nt)
      op[nt * 16 + lr] = (e[nt] - mean) * rstd * g_[nt] + b_[nt];
  }
}

// ---------------------------------------------------------------------------
extern "C" void kernel_launch(void* const* d_in, const int* in_sizes, int n_in,
                              void* d_out, int out_size, void* d_ws, size_t ws_size,
                              hipStream_t stream) {
  const float* x     = (const float*)d_in[0];
  const float* W1    = (const float*)d_in[1];
  const float* b1    = (const float*)d_in[2];
  const float* W2    = (const float*)d_in[3];
  const float* b2    = (const float*)d_in[4];
  const float* embed = (const float*)d_in[5];
  const float* g     = (const float*)d_in[6];
  const float* bet   = (const float*)d_in[7];
  float* out = (float*)d_out;

  char* ws = (char*)d_ws;
  bf16_t* w1t  = (bf16_t*)(ws);                          // 256 x 1024  (512 KB)
  bf16_t* w2t  = (bf16_t*)(ws + (size_t)512 * 1024);     // 4096 x 256  (2 MB)
  bf16_t* embT = (bf16_t*)(ws + (size_t)2560 * 1024);    // 64 x 4096   (512 KB)
  bf16_t* hbuf = (bf16_t*)(ws + (size_t)3072 * 1024);    // 16384 x 256 (8 MB)

  cvt_transpose_bf16<<<(TASK_EMB * HID + 255) / 256, 256, 0, stream>>>(W1, w1t, TASK_EMB, HID);
  cvt_transpose_bf16<<<(HID * N_EMBS + 255) / 256, 256, 0, stream>>>(W2, w2t, HID, N_EMBS);
  cvt_transpose_bf16<<<(N_EMBS * CODE_DIM + 255) / 256, 256, 0, stream>>>(embed, embT, N_EMBS, CODE_DIM);

  // 1024 row-tiles, 8 waves/block -> 128 blocks
  gemm1_silu<<<BATCH / 16 / 8, 256, 0, stream>>>(x, w1t, b1, hbuf);
  fused_softmax_code<<<BATCH / 16 / 8, 256, 0, stream>>>(hbuf, w2t, b2, embT, g, bet, out);
}
